// Scaled_Dot_Product_Attention_68178310856820
// MI455X (gfx1250) — matrix operand order repaired and checked by reading, not compiled, not run on hardware
//
#include <hip/hip_runtime.h>
#include <hip/hip_bf16.h>

typedef __attribute__((ext_vector_type(16))) _Float16 v16h;
typedef __attribute__((ext_vector_type(8)))  _Float16 v8h;
typedef __attribute__((ext_vector_type(4)))  _Float16 v4h;
typedef __attribute__((ext_vector_type(8)))  float    v8f;
typedef __attribute__((ext_vector_type(4)))  int      v4i;

#define BATCH  2
#define HEADS  16
#define SEQ    2048
#define DKDIM  64
#define QTILE  16     // query rows per wave
#define KTILE  32     // keys per inner iteration
#define WAVES  4      // waves per block
#define NBH    (BATCH * HEADS)
#define NELEM  ((size_t)NBH * SEQ * DKDIM)          // 4,194,304 per tensor

#if __has_builtin(__builtin_amdgcn_global_load_async_to_lds_b128)
#define HAVE_ASYNC_LDS 1
#else
#define HAVE_ASYNC_LDS 0
#endif

static __device__ __forceinline__ v16h cat16(v8h lo, v8h hi) {
    return __builtin_shufflevector(lo, hi, 0, 1, 2, 3, 4, 5, 6, 7,
                                           8, 9, 10, 11, 12, 13, 14, 15);
}

// copy 8 halves (16B) global -> LDS, async if available
static __device__ __forceinline__ void stage8(const _Float16* g, _Float16* l) {
#if HAVE_ASYNC_LDS
    __builtin_amdgcn_global_load_async_to_lds_b128(
        (__attribute__((address_space(1))) v4i*)g,
        (__attribute__((address_space(3))) v4i*)l, 0, 0);
#else
    *(v8h*)l = *(const v8h*)g;
#endif
}

static __device__ __forceinline__ void wait_stage() {
#if HAVE_ASYNC_LDS
    asm volatile("s_wait_asynccnt 0x0" ::: "memory");
#endif
}

// ---------------------------------------------------------------------------
// Pre-pass 1: Q*scale -> f16, K -> f16 (same [bh][s][d] layout)
// ---------------------------------------------------------------------------
__global__ __launch_bounds__(256)
void prep_qk(const float* __restrict__ Q, const float* __restrict__ K,
             _Float16* __restrict__ Qh, _Float16* __restrict__ Kh)
{
    const size_t i = ((size_t)blockIdx.x * blockDim.x + threadIdx.x) * 4;
    const float4 q4 = *(const float4*)(Q + i);
    const float4 k4 = *(const float4*)(K + i);
    v4h qh, kh;
    qh[0] = (_Float16)(q4.x * 0.125f);
    qh[1] = (_Float16)(q4.y * 0.125f);
    qh[2] = (_Float16)(q4.z * 0.125f);
    qh[3] = (_Float16)(q4.w * 0.125f);
    kh[0] = (_Float16)k4.x;
    kh[1] = (_Float16)k4.y;
    kh[2] = (_Float16)k4.z;
    kh[3] = (_Float16)k4.w;
    *(v4h*)(Qh + i) = qh;
    *(v4h*)(Kh + i) = kh;
}

// ---------------------------------------------------------------------------
// Pre-pass 2: V -> f16 transposed  Vt[bh][d][s] = V[bh][s][d]
// ---------------------------------------------------------------------------
__global__ __launch_bounds__(256)
void prep_vT(const float* __restrict__ V, _Float16* __restrict__ Vt)
{
    const size_t idx = (size_t)blockIdx.x * blockDim.x + threadIdx.x;
    const int s  = (int)(idx % (SEQ / 4)) * 4;
    const int d  = (int)((idx / (SEQ / 4)) % DKDIM);
    const int bh = (int)(idx / ((size_t)(SEQ / 4) * DKDIM));
    const float* vp = V + ((size_t)bh * SEQ + s) * DKDIM + d;
    v4h ov;
    #pragma unroll
    for (int j = 0; j < 4; ++j)
        ov[j] = (_Float16)vp[(size_t)j * DKDIM];
    *(v4h*)(Vt + ((size_t)bh * DKDIM + d) * SEQ + s) = ov;
}

// ---------------------------------------------------------------------------
// Flash attention: async double-buffered K/V tiles in LDS, f16 WMMA
// ---------------------------------------------------------------------------
__global__ __launch_bounds__(WAVES * 32)
void fa_fwd_pre(const _Float16* __restrict__ Qh,
                const _Float16* __restrict__ Kh,
                const _Float16* __restrict__ Vt,
                float* __restrict__ O)
{
    __shared__ _Float16 ldsK[2][KTILE][DKDIM];      // 2 x 4 KB
    __shared__ _Float16 ldsV[2][DKDIM][KTILE];      // 2 x 4 KB
    __shared__ _Float16 ldsP[WAVES][QTILE][KTILE];  // 4 KB (P transpose)

    const int tid  = threadIdx.x;
    const int wave = tid >> 5;
    const int lane = tid & 31;
    const int half = lane >> 4;
    const int m    = lane & 15;

    const int qblocks = SEQ / (QTILE * WAVES);
    const int bh      = blockIdx.x / qblocks;
    const int qblk    = blockIdx.x % qblocks;
    const int qbase   = qblk * (QTILE * WAVES) + wave * QTILE;

    const _Float16* q  = Qh + (size_t)bh * SEQ * DKDIM;
    const _Float16* k  = Kh + (size_t)bh * SEQ * DKDIM;
    const _Float16* vt = Vt + (size_t)bh * DKDIM * SEQ;
    float*          o  = O  + (size_t)bh * SEQ * DKDIM;

    // stage one 32-key tile (K: contiguous 4 KB; V: 64 dim-rows x 64 B)
    auto stage_tile = [&](int tt) {
        const int ktt = tt * KTILE;
        const int buf = tt & 1;
        const _Float16* gk = k + (size_t)ktt * DKDIM;
        _Float16* lk = &ldsK[buf][0][0];
        stage8(gk + tid * 8,        lk + tid * 8);
        stage8(gk + 1024 + tid * 8, lk + 1024 + tid * 8);
        const _Float16* gv = vt + (size_t)(tid >> 1) * SEQ + ktt + (tid & 1) * 16;
        _Float16* lv = &ldsV[buf][tid >> 1][0] + (tid & 1) * 16;
        stage8(gv,     lv);
        stage8(gv + 8, lv + 8);
    };

    // Q as two f16 A-fragments (K=0..31, K=32..63); scale pre-folded
    v16h a0, a1;
    {
        const _Float16* qp = q + (size_t)(qbase + m) * DKDIM;
        a0 = cat16(*(const v8h*)(qp +      16 * half),
                   *(const v8h*)(qp + 8 +  16 * half));
        a1 = cat16(*(const v8h*)(qp + 32 + 16 * half),
                   *(const v8h*)(qp + 40 + 16 * half));
    }

    float rowmax[8], rowsum[8];
    v8f acc[4];
    #pragma unroll
    for (int r = 0; r < 8; ++r) { rowmax[r] = -1e30f; rowsum[r] = 0.0f; }
    #pragma unroll
    for (int d = 0; d < 4; ++d) acc[d] = (v8f){};

    const int nk  = qbase + QTILE;        // this wave's causal key bound
    const int ntB = qblk * 2 + 2;         // block-uniform tile count

    stage_tile(0);

    for (int t = 0; t < ntB; ++t) {
        const int kt  = t * KTILE;
        const int buf = t & 1;

        wait_stage();            // own async copies for tile t complete
        __syncthreads();         // all staging(t) done; all compute(t-1) done

        if (t + 1 < ntB) {       // overlap next tile's DMA with compute(t)
            stage_tile(t + 1);
            __builtin_prefetch(k + (size_t)(kt + 2 * KTILE) * DKDIM, 0, 0);
        }

        if (kt < nk) {
            const _Float16* lk = &ldsK[buf][0][0];
            const _Float16* lv = &ldsV[buf][0][0];

            // ---- K tile B-fragments from LDS
            v16h bk[2][2];
            #pragma unroll
            for (int g = 0; g < 2; ++g) {
                const _Float16* kp = lk + (g * 16 + m) * DKDIM + 16 * half;
                #pragma unroll
                for (int w = 0; w < 2; ++w)
                    bk[g][w] = cat16(*(const v8h*)(kp + 32 * w),
                                     *(const v8h*)(kp + 32 * w + 8));
            }

            // ---- scores
            v8f c0 = (v8f){}, c1 = (v8f){};
            c0 = __builtin_amdgcn_wmma_f32_16x16x32_f16(false, a0, false, bk[0][0],
                                                        (short)0, c0, false, false);
            c0 = __builtin_amdgcn_wmma_f32_16x16x32_f16(false, a1, false, bk[0][1],
                                                        (short)0, c0, false, false);
            c1 = __builtin_amdgcn_wmma_f32_16x16x32_f16(false, a0, false, bk[1][0],
                                                        (short)0, c1, false, false);
            c1 = __builtin_amdgcn_wmma_f32_16x16x32_f16(false, a1, false, bk[1][1],
                                                        (short)0, c1, false, false);

            // ---- causal mask (analytic)
            const int key0 = kt + m;
            const int key1 = kt + 16 + m;
            #pragma unroll
            for (int r = 0; r < 8; ++r) {
                const int row = qbase + 8 * half + r;
                if (key0 > row) c0[r] = -1e30f;
                if (key1 > row) c1[r] = -1e30f;
            }

            // ---- online softmax (reductions within 16-lane halves)
            float p0[8], p1[8], corr[8];
            #pragma unroll
            for (int r = 0; r < 8; ++r) {
                float tm = fmaxf(c0[r], c1[r]);
                #pragma unroll
                for (int s = 8; s >= 1; s >>= 1)
                    tm = fmaxf(tm, __shfl_xor(tm, s));
                const float mnew = fmaxf(rowmax[r], tm);
                corr[r]   = __expf(rowmax[r] - mnew);
                rowmax[r] = mnew;
                p0[r] = __expf(c0[r] - mnew);
                p1[r] = __expf(c1[r] - mnew);
                float ts = p0[r] + p1[r];
                #pragma unroll
                for (int s = 8; s >= 1; s >>= 1)
                    ts += __shfl_xor(ts, s);
                rowsum[r] = rowsum[r] * corr[r] + ts;
            }
            #pragma unroll
            for (int d = 0; d < 4; ++d) {
                #pragma unroll
                for (int r = 0; r < 8; ++r)
                    acc[d][r] *= corr[r];
            }

            // ---- transpose P (C layout -> A layout) via wave-private LDS
            _Float16* pb = &ldsP[wave][0][0];
            #pragma unroll
            for (int r = 0; r < 8; ++r) {
                const int row = 8 * half + r;
                pb[row * KTILE + m]      = (_Float16)p0[r];
                pb[row * KTILE + 16 + m] = (_Float16)p1[r];
            }
            asm volatile("s_wait_dscnt 0" ::: "memory");
            v16h pa;
            {
                const _Float16* pr = pb + m * KTILE;
                pa = cat16(*(const v8h*)(pr +      16 * half),
                           *(const v8h*)(pr + 8 +  16 * half));
            }

            // ---- O += P . V from LDS (contiguous per-lane rows of Vt tile)
            #pragma unroll
            for (int d = 0; d < 4; ++d) {
                const _Float16* vp = lv + (16 * d + m) * KTILE + 16 * half;
                v16h bv = cat16(*(const v8h*)vp, *(const v8h*)(vp + 8));
                acc[d] = __builtin_amdgcn_wmma_f32_16x16x32_f16(false, pa, false, bv,
                                                                (short)0, acc[d],
                                                                false, false);
            }
        }
    }

    // ---- normalize and store
    #pragma unroll
    for (int r = 0; r < 8; ++r) {
        const int row = qbase + 8 * half + r;
        const float inv = 1.0f / rowsum[r];
        #pragma unroll
        for (int d = 0; d < 4; ++d)
            o[(size_t)row * DKDIM + 16 * d + m] = acc[d][r] * inv;
    }
}

// ---------------------------------------------------------------------------
// Fallback: direct-f32 version (used only if d_ws is too small)
// ---------------------------------------------------------------------------
__global__ __launch_bounds__(WAVES * 32)
void fa_fwd_direct(const float* __restrict__ Q, const float* __restrict__ K,
                   const float* __restrict__ V, float* __restrict__ O)
{
    __shared__ _Float16 ldsP[WAVES][QTILE][KTILE];

    const int tid  = threadIdx.x;
    const int wave = tid >> 5;
    const int lane = tid & 31;
    const int half = lane >> 4;
    const int m    = lane & 15;

    const int qblocks = SEQ / (QTILE * WAVES);
    const int bh      = blockIdx.x / qblocks;
    const int qblk    = blockIdx.x % qblocks;
    const int qbase   = qblk * (QTILE * WAVES) + wave * QTILE;

    const size_t headoff = (size_t)bh * SEQ * DKDIM;
    const float* q = Q + headoff;
    const float* k = K + headoff;
    const float* v = V + headoff;
    float*       o = O + headoff;

    const float scale = 0.125f;
    v16h a0, a1;
    {
        const float* qp = q + (size_t)(qbase + m) * DKDIM;
        #pragma unroll
        for (int j = 0; j < 8; ++j) {
            a0[j]     = (_Float16)(qp[     16 * half + j] * scale);
            a0[8 + j] = (_Float16)(qp[8 +  16 * half + j] * scale);
            a1[j]     = (_Float16)(qp[32 + 16 * half + j] * scale);
            a1[8 + j] = (_Float16)(qp[40 + 16 * half + j] * scale);
        }
    }

    float rowmax[8], rowsum[8];
    v8f acc[4];
    #pragma unroll
    for (int r = 0; r < 8; ++r) { rowmax[r] = -1e30f; rowsum[r] = 0.0f; }
    #pragma unroll
    for (int d = 0; d < 4; ++d) acc[d] = (v8f){};

    const int nk     = qbase + QTILE;
    const int ntiles = (nk + KTILE - 1) / KTILE;

    for (int t = 0; t < ntiles; ++t) {
        const int kt = t * KTILE;

        v16h bk[2][2];
        #pragma unroll
        for (int g = 0; g < 2; ++g) {
            const float* kp = k + (size_t)(kt + g * 16 + m) * DKDIM + 16 * half;
            #pragma unroll
            for (int w = 0; w < 2; ++w) {
                #pragma unroll
                for (int j = 0; j < 16; ++j)
                    bk[g][w][j] = (_Float16)kp[32 * w + j];
            }
        }

        v8f c0 = (v8f){}, c1 = (v8f){};
        c0 = __builtin_amdgcn_wmma_f32_16x16x32_f16(false, a0, false, bk[0][0],
                                                    (short)0, c0, false, false);
        c0 = __builtin_amdgcn_wmma_f32_16x16x32_f16(false, a1, false, bk[0][1],
                                                    (short)0, c0, false, false);
        c1 = __builtin_amdgcn_wmma_f32_16x16x32_f16(false, a0, false, bk[1][0],
                                                    (short)0, c1, false, false);
        c1 = __builtin_amdgcn_wmma_f32_16x16x32_f16(false, a1, false, bk[1][1],
                                                    (short)0, c1, false, false);

        const int key0 = kt + m;
        const int key1 = kt + 16 + m;
        #pragma unroll
        for (int r = 0; r < 8; ++r) {
            const int row = qbase + 8 * half + r;
            if (key0 > row) c0[r] = -1e30f;
            if (key1 > row) c1[r] = -1e30f;
        }

        float p0[8], p1[8], corr[8];
        #pragma unroll
        for (int r = 0; r < 8; ++r) {
            float tm = fmaxf(c0[r], c1[r]);
            #pragma unroll
            for (int s = 8; s >= 1; s >>= 1)
                tm = fmaxf(tm, __shfl_xor(tm, s));
            const float mnew = fmaxf(rowmax[r], tm);
            corr[r]   = __expf(rowmax[r] - mnew);
            rowmax[r] = mnew;
            p0[r] = __expf(c0[r] - mnew);
            p1[r] = __expf(c1[r] - mnew);
            float ts = p0[r] + p1[r];
            #pragma unroll
            for (int s = 8; s >= 1; s >>= 1)
                ts += __shfl_xor(ts, s);
            rowsum[r] = rowsum[r] * corr[r] + ts;
        }
        #pragma unroll
        for (int d = 0; d < 4; ++d) {
            #pragma unroll
            for (int r = 0; r < 8; ++r)
                acc[d][r] *= corr[r];
        }

        _Float16* pb = &ldsP[wave][0][0];
        #pragma unroll
        for (int r = 0; r < 8; ++r) {
            const int row = 8 * half + r;
            pb[row * KTILE + m]      = (_Float16)p0[r];
            pb[row * KTILE + 16 + m] = (_Float16)p1[r];
        }
        asm volatile("s_wait_dscnt 0" ::: "memory");
        v16h pa;
        {
            const _Float16* pr = pb + m * KTILE;
            pa = cat16(*(const v8h*)(pr +      16 * half),
                       *(const v8h*)(pr + 8 +  16 * half));
        }

        #pragma unroll
        for (int d = 0; d < 4; ++d) {
            v16h bv;
            const float* vp = v + (size_t)(kt + 16 * half) * DKDIM + 16 * d + m;
            #pragma unroll
            for (int j = 0; j < 16; ++j)
                bv[j] = (_Float16)vp[(size_t)j * DKDIM];
            acc[d] = __builtin_amdgcn_wmma_f32_16x16x32_f16(false, pa, false, bv,
                                                            (short)0, acc[d],
                                                            false, false);
        }
    }

    #pragma unroll
    for (int r = 0; r < 8; ++r) {
        const int row = qbase + 8 * half + r;
        const float inv = 1.0f / rowsum[r];
        #pragma unroll
        for (int d = 0; d < 4; ++d)
            o[(size_t)row * DKDIM + 16 * d + m] = acc[d][r] * inv;
    }
}

// ---------------------------------------------------------------------------
extern "C" void kernel_launch(void* const* d_in, const int* in_sizes, int n_in,
                              void* d_out, int out_size, void* d_ws, size_t ws_size,
                              hipStream_t stream)
{
    const float* Q = (const float*)d_in[0];
    const float* K = (const float*)d_in[1];
    const float* V = (const float*)d_in[2];
    float*       O = (float*)d_out;

    const size_t bytesPerHalfTensor = NELEM * sizeof(_Float16);   // 8 MB
    const size_t needed = 3 * bytesPerHalfTensor;                 // 24 MB

    dim3 faBlock(WAVES * 32);
    dim3 faGrid(NBH * (SEQ / (QTILE * WAVES)));

    if (ws_size >= needed) {
        _Float16* Qh = (_Float16*)d_ws;
        _Float16* Kh = (_Float16*)((char*)d_ws + bytesPerHalfTensor);
        _Float16* Vt = (_Float16*)((char*)d_ws + 2 * bytesPerHalfTensor);

        hipLaunchKernelGGL(prep_qk, dim3((unsigned)(NELEM / 4 / 256)), dim3(256),
                           0, stream, Q, K, Qh, Kh);
        hipLaunchKernelGGL(prep_vT, dim3((unsigned)(NELEM / 4 / 256)), dim3(256),
                           0, stream, V, Vt);
        hipLaunchKernelGGL(fa_fwd_pre, faGrid, faBlock, 0, stream, Qh, Kh, Vt, O);
    } else {
        hipLaunchKernelGGL(fa_fwd_direct, faGrid, faBlock, 0, stream, Q, K, V, O);
    }
}
